// CrossEncoderReranker_70205535420853
// MI455X (gfx1250) — matrix operand order repaired and checked by reading, not compiled, not run on hardware
//
#include <hip/hip_runtime.h>
#include <hip/hip_bf16.h>

// ---------------- model constants ----------------
#define VV   16384
#define DD   384
#define SS   512
#define BB   128
#define HH   6
#define HD   64
#define DIN  768
#define DFF  1536
#define NM   6
#define NA   2
#define EPS  1e-05f
#define NTOK (BB * SS)          // 65536 token rows

typedef unsigned short u16;
typedef __attribute__((ext_vector_type(16))) __bf16 bf16x16;
typedef __attribute__((ext_vector_type(16))) u16    u16x16;
typedef __attribute__((ext_vector_type(8)))  u16    u16x8;
typedef __attribute__((ext_vector_type(8)))  float  f32x8;
typedef __attribute__((ext_vector_type(4)))  int    i32x4;

// address-space qualified 16B vector types for the async-LDS builtin
typedef __attribute__((address_space(1))) i32x4 gas_i32x4;   // global
typedef __attribute__((address_space(3))) i32x4 las_i32x4;   // LDS

// fp32 -> bf16 (round to nearest even)
__device__ __forceinline__ u16 f2bf(float f) {
    unsigned int u = __builtin_bit_cast(unsigned int, f);
    u = (u + 0x7FFFu + ((u >> 16) & 1u)) >> 16;
    return (u16)u;
}

// ---- async global->LDS 128-bit copy (CDNA5), with safe fallback ----
#if __has_builtin(__builtin_amdgcn_global_load_async_to_lds_b128) && \
    __has_builtin(__builtin_amdgcn_s_wait_asynccnt)
#define HAVE_ASYNC_LDS 1
#else
#define HAVE_ASYNC_LDS 0
#endif

__device__ __forceinline__ void copy_b128_g2l(const u16* g, u16* l) {
#if HAVE_ASYNC_LDS
    __builtin_amdgcn_global_load_async_to_lds_b128(
        (gas_i32x4*)(u16*)g,      // C-style cast: const drop + addrspacecast to AS1
        (las_i32x4*)l,            // addrspacecast generic LDS pointer to AS3
        0, 0);
#else
    *(u16x8*)l = *(const u16x8*)g;
#endif
}
__device__ __forceinline__ void wait_async_copies() {
#if HAVE_ASYNC_LDS
    __builtin_amdgcn_s_wait_asynccnt(0);
#endif
}

// ---- WMMA fragment loaders (LDS -> VGPR) per CDNA5 16-bit layouts ----
// A: 16x32 (MxK). lanes0-15: M=lane, K {k0..k0+7, k0+16..k0+23}
//                 lanes16-31: M=lane-16, K {k0+8..k0+15, k0+24..k0+31}
__device__ __forceinline__ bf16x16 load_frag_a(const u16* a, int ld, int k0) {
    int lane = threadIdx.x & 31;
    int half = lane >> 4;
    int m    = lane & 15;
    const u16* p = a + m * ld;
    int b1 = k0 + half * 8;
    int b2 = k0 + 16 + half * 8;
    u16x16 r;
#pragma unroll
    for (int i = 0; i < 8; ++i) { r[i] = p[b1 + i]; r[8 + i] = p[b2 + i]; }
    return __builtin_bit_cast(bf16x16, r);
}

// B: 32x16 (KxN), loaded from B^T tile in LDS: bt[n][k], row stride ld.
// lanes0-15: N=lane, K {k0..k0+7, k0+16..k0+23} ; lanes16-31: N=lane-16, K {k0+8..k0+15, k0+24..k0+31}
__device__ __forceinline__ bf16x16 load_frag_b(const u16* bt, int ld, int k0) {
    int lane = threadIdx.x & 31;
    int half = lane >> 4;
    int n    = lane & 15;
    const u16* p = bt + n * ld + k0;
    u16x16 r;
#pragma unroll
    for (int i = 0; i < 8; ++i) { r[i] = p[half * 8 + i]; r[8 + i] = p[16 + half * 8 + i]; }
    return __builtin_bit_cast(bf16x16, r);
}

// ---------------- weight convert + transpose: W[K][N] f32 -> Wt[N][K] bf16 ----------------
__global__ __launch_bounds__(256)
void cer_wt_kernel(const float* __restrict__ W, u16* __restrict__ Wt, int K, int N) {
    int idx = blockIdx.x * 256 + threadIdx.x;
    if (idx >= K * N) return;
    int n = idx / K, k = idx - n * K;           // coalesced writes along K
    Wt[(size_t)n * K + k] = f2bf(W[(size_t)k * N + n]);
}

// ---------------- embedding gather + pos add ----------------
__global__ __launch_bounds__(128)
void cer_embed_kernel(const int* __restrict__ ids, const float* __restrict__ emb,
                      const float* __restrict__ pos, float* __restrict__ x) {
    int row = blockIdx.x;               // b*S + s
    int s   = row & (SS - 1);
    int id  = ids[row];
    const float* e = emb + (size_t)id * DD;
    const float* p = pos + (size_t)s * DD;
    float* o = x + (size_t)row * DD;
    for (int d = threadIdx.x; d < DD; d += 128) o[d] = e[d] + p[d];
}

// ---------------- LayerNorm (optional residual, optional f32/bf16 outputs) ----------------
__global__ __launch_bounds__(128)
void cer_ln_kernel(const float* __restrict__ in, const float* __restrict__ res,
                   const float* __restrict__ w, const float* __restrict__ b,
                   float* __restrict__ out, u16* __restrict__ outb) {
    int row = blockIdx.x;
    int tid = threadIdx.x;
    __shared__ float red[128];
    const float* pi = in + (size_t)row * DD;
    const float* pr = res ? res + (size_t)row * DD : nullptr;
    float v[3];
    float s = 0.f;
#pragma unroll
    for (int i = 0; i < 3; ++i) {
        int d = tid + i * 128;
        float t = pi[d];
        if (pr) t += pr[d];
        v[i] = t; s += t;
    }
    red[tid] = s; __syncthreads();
    for (int off = 64; off > 0; off >>= 1) {
        if (tid < off) red[tid] += red[tid + off];
        __syncthreads();
    }
    float mu = red[0] * (1.0f / DD);
    __syncthreads();
    float q = 0.f;
#pragma unroll
    for (int i = 0; i < 3; ++i) { float d = v[i] - mu; q += d * d; }
    red[tid] = q; __syncthreads();
    for (int off = 64; off > 0; off >>= 1) {
        if (tid < off) red[tid] += red[tid + off];
        __syncthreads();
    }
    float rstd = rsqrtf(red[0] * (1.0f / DD) + EPS);
#pragma unroll
    for (int i = 0; i < 3; ++i) {
        int d = tid + i * 128;
        float y = (v[i] - mu) * rstd * w[d] + b[d];
        if (out)  out [(size_t)row * DD + d] = y;
        if (outb) outb[(size_t)row * DD + d] = f2bf(y);
    }
}

// ---------------- bf16-WMMA GEMM ----------------
// A[M][K] bf16, Wt[N][K] bf16 -> val = act(A*W + bias) (+Res f32)
// optional outputs: C (f32), Cb (bf16).  Tiles 64x64x64, 4 waves/block.
#define ACT_NONE 0
#define ACT_SILU 1
#define ACT_RELU 2

__global__ __launch_bounds__(128)
void cer_gemm_kernel(const u16* __restrict__ A, const u16* __restrict__ Wt,
                     const float* __restrict__ bias, const float* __restrict__ Res,
                     float* __restrict__ C, u16* __restrict__ Cb,
                     int M, int K, int N, int act) {
    __shared__ u16 a_lds[64][64];    // A tile bf16
    __shared__ u16 bt_lds[64][64];   // W^T tile bf16: bt[n][k]
    int tid  = threadIdx.x;
    int wave = tid >> 5;
    int row_base = blockIdx.x * 64;
    int col_base = blockIdx.y * 64;

    f32x8 acc[4];
#pragma unroll
    for (int nt = 0; nt < 4; ++nt) acc[nt] = (f32x8){0.f,0.f,0.f,0.f,0.f,0.f,0.f,0.f};

    for (int k0 = 0; k0 < K; k0 += 64) {
        // 64x64 bf16 tiles: 512 x b128 copies each, 4 per thread
        for (int i = tid; i < 64 * 8; i += 128) {
            int r = i >> 3, c = (i & 7) * 8;
            copy_b128_g2l(A + (size_t)(row_base + r) * K + k0 + c, &a_lds[r][c]);
        }
        for (int i = tid; i < 64 * 8; i += 128) {
            int n = i >> 3, c = (i & 7) * 8;
            copy_b128_g2l(Wt + (size_t)(col_base + n) * K + k0 + c, &bt_lds[n][c]);
        }
        wait_async_copies();
        __syncthreads();
#pragma unroll
        for (int kk = 0; kk < 64; kk += 32) {
            bf16x16 af = load_frag_a(&a_lds[wave * 16][0], 64, kk);
#pragma unroll
            for (int nt = 0; nt < 4; ++nt) {
                bf16x16 bfr = load_frag_b(&bt_lds[nt * 16][0], 64, kk);
                acc[nt] = __builtin_amdgcn_wmma_f32_16x16x32_bf16(
                    false, af, false, bfr, (short)0, acc[nt], false, false);
            }
        }
        __syncthreads();
    }

    int lane = tid & 31, half = lane >> 4, nloc = lane & 15;
#pragma unroll
    for (int nt = 0; nt < 4; ++nt) {
#pragma unroll
        for (int v = 0; v < 8; ++v) {
            int r = row_base + wave * 16 + v + half * 8;
            int n = col_base + nt * 16 + nloc;
            float val = acc[nt][v] + bias[n];
            if (act == ACT_SILU)      val = val / (1.0f + __expf(-val));
            else if (act == ACT_RELU) val = fmaxf(val, 0.0f);
            if (Res) val += Res[(size_t)r * N + n];
            if (C)  C [(size_t)r * N + n] = val;
            if (Cb) Cb[(size_t)r * N + n] = f2bf(val);
        }
    }
}

// ---------------- fused flash attention (bf16 qkv in, bf16 O out) ----------------
__global__ __launch_bounds__(128)
void cer_attn_kernel(const u16* __restrict__ qkv, u16* __restrict__ ob) {
    __shared__ u16  q_lds[64][64];    // Q tile [row][dim]
    __shared__ u16  kt_lds[64][64];   // K tile [key][dim] == B^T for QK^T
    __shared__ u16  vt_lds[64][64];   // V^T tile [dim][key] == B^T for P*V
    __shared__ float s_lds[64][64];   // score tile f32
    __shared__ u16  p_lds[64][64];    // prob tile bf16
    __shared__ float mrow[64], lrow[64], arow[64];
    __shared__ float pmax[64][2], psum[64][2];

    int qt = blockIdx.x, h = blockIdx.y, b = blockIdx.z;
    int tid = threadIdx.x, wave = tid >> 5, lane = tid & 31;
    int half = lane >> 4, nloc = lane & 15;
    const float scale = 0.125f;       // 1/sqrt(64)
    const int LDQ = 3 * DD;

    size_t rowb = (size_t)b * SS;
    int q0 = qt * 64;

    // load Q tile (vector copies)
    for (int i = tid; i < 64 * 8; i += 128) {
        int r = i >> 3, c = (i & 7) * 8;
        copy_b128_g2l(qkv + (rowb + q0 + r) * LDQ + h * HD + c, &q_lds[r][c]);
    }
    if (tid < 64) { mrow[tid] = -1e30f; lrow[tid] = 0.f; }

    f32x8 acc[4];
#pragma unroll
    for (int nt = 0; nt < 4; ++nt) acc[nt] = (f32x8){0.f,0.f,0.f,0.f,0.f,0.f,0.f,0.f};
    wait_async_copies();
    __syncthreads();

    for (int kt = 0; kt < SS / 64; ++kt) {
        // K tile: direct vector copy; V tile: transposed scalar copy
        for (int i = tid; i < 64 * 8; i += 128) {
            int key = i >> 3, c = (i & 7) * 8;
            copy_b128_g2l(qkv + (rowb + kt * 64 + key) * LDQ + DD + h * HD + c,
                          &kt_lds[key][c]);
        }
        for (int i = tid; i < 64 * 64; i += 128) {
            int key = i >> 6, d = i & 63;
            vt_lds[d][key] = qkv[(rowb + kt * 64 + key) * LDQ + 2 * DD + h * HD + d];
        }
        wait_async_copies();
        __syncthreads();

        // S = scale * Q K^T
        f32x8 sa[4];
#pragma unroll
        for (int nt = 0; nt < 4; ++nt) sa[nt] = (f32x8){0.f,0.f,0.f,0.f,0.f,0.f,0.f,0.f};
#pragma unroll
        for (int kk = 0; kk < 64; kk += 32) {
            bf16x16 qf = load_frag_a(&q_lds[wave * 16][0], 64, kk);
#pragma unroll
            for (int nt = 0; nt < 4; ++nt) {
                bf16x16 kf = load_frag_b(&kt_lds[nt * 16][0], 64, kk);
                sa[nt] = __builtin_amdgcn_wmma_f32_16x16x32_bf16(
                    false, qf, false, kf, (short)0, sa[nt], false, false);
            }
        }
#pragma unroll
        for (int nt = 0; nt < 4; ++nt)
#pragma unroll
            for (int v = 0; v < 8; ++v)
                s_lds[wave * 16 + v + half * 8][nt * 16 + nloc] = sa[nt][v] * scale;
        __syncthreads();

        // online softmax over this 64-key tile
        {
            int r = tid >> 1, hh = tid & 1, base = hh * 32;
            float mx = -1e30f;
#pragma unroll
            for (int j = 0; j < 32; ++j) mx = fmaxf(mx, s_lds[r][base + j]);
            pmax[r][hh] = mx;
            __syncthreads();
            if (hh == 0) {
                float m_new = fmaxf(mrow[r], fmaxf(pmax[r][0], pmax[r][1]));
                arow[r] = __expf(mrow[r] - m_new);
                mrow[r] = m_new;
            }
            __syncthreads();
            float m_new = mrow[r], s = 0.f;
#pragma unroll
            for (int j = 0; j < 32; ++j) {
                float p = __expf(s_lds[r][base + j] - m_new);
                p_lds[r][base + j] = f2bf(p);
                s += p;
            }
            psum[r][hh] = s;
            __syncthreads();
            if (hh == 0) lrow[r] = lrow[r] * arow[r] + psum[r][0] + psum[r][1];
            __syncthreads();
        }

        // rescale O, then O += P * V
#pragma unroll
        for (int nt = 0; nt < 4; ++nt)
#pragma unroll
            for (int v = 0; v < 8; ++v)
                acc[nt][v] *= arow[wave * 16 + v + half * 8];
#pragma unroll
        for (int kk = 0; kk < 64; kk += 32) {
            bf16x16 pf = load_frag_a(&p_lds[wave * 16][0], 64, kk);
#pragma unroll
            for (int nt = 0; nt < 4; ++nt) {
                bf16x16 vf = load_frag_b(&vt_lds[nt * 16][0], 64, kk);
                acc[nt] = __builtin_amdgcn_wmma_f32_16x16x32_bf16(
                    false, pf, false, vf, (short)0, acc[nt], false, false);
            }
        }
        __syncthreads();
    }

    // normalize and write bf16 O into (B*S, D) layout at head offset
#pragma unroll
    for (int nt = 0; nt < 4; ++nt)
#pragma unroll
        for (int v = 0; v < 8; ++v) {
            int rl = wave * 16 + v + half * 8;
            int r  = q0 + rl;
            int d  = nt * 16 + nloc;
            ob[(rowb + r) * DD + h * HD + d] = f2bf(acc[nt][v] / lrow[rl]);
        }
}

// ---------------- CLS head: relu(cls @ W1 + b1) @ W2 + b2 ----------------
__global__ __launch_bounds__(128)
void cer_head_kernel(const float* __restrict__ xf, const float* __restrict__ w1,
                     const float* __restrict__ b1, const float* __restrict__ w2,
                     const float* __restrict__ b2, float* __restrict__ out) {
    int b = blockIdx.x, t = threadIdx.x;
    __shared__ float red[128];
    const float* cls = xf + (size_t)b * SS * DD;   // row b*S + 0
    float a = 0.f;
    for (int k = 0; k < DD; ++k) a += cls[k] * w1[k * 128 + t];
    a = fmaxf(a + b1[t], 0.f);
    red[t] = a * w2[t];
    __syncthreads();
    for (int off = 64; off > 0; off >>= 1) {
        if (t < off) red[t] += red[t + off];
        __syncthreads();
    }
    if (t == 0) out[b] = red[0] + b2[0];
}

// ---------------- host launcher ----------------
extern "C" void kernel_launch(void* const* d_in, const int* in_sizes, int n_in,
                              void* d_out, int out_size, void* d_ws, size_t ws_size,
                              hipStream_t stream) {
    const int*   ids      = (const int*)  d_in[0];
    const float* emb      = (const float*)d_in[1];
    const float* pos_emb  = (const float*)d_in[2];
    const float* m_ln_w   = (const float*)d_in[3];
    const float* m_ln_b   = (const float*)d_in[4];
    const float* m_W1     = (const float*)d_in[5];
    const float* m_b1     = (const float*)d_in[6];
    const float* m_W2     = (const float*)d_in[7];
    const float* m_b2     = (const float*)d_in[8];
    const float* a_qkv_w  = (const float*)d_in[9];
    const float* a_qkv_b  = (const float*)d_in[10];
    const float* a_out_w  = (const float*)d_in[11];
    const float* a_out_b  = (const float*)d_in[12];
    const float* a_ln1_w  = (const float*)d_in[13];
    const float* a_ln1_b  = (const float*)d_in[14];
    const float* a_ln2_w  = (const float*)d_in[15];
    const float* a_ln2_b  = (const float*)d_in[16];
    const float* a_ff_w1  = (const float*)d_in[17];
    const float* a_ff_b1  = (const float*)d_in[18];
    const float* a_ff_w2  = (const float*)d_in[19];
    const float* a_ff_b2  = (const float*)d_in[20];
    const float* fn_w     = (const float*)d_in[21];
    const float* fn_b     = (const float*)d_in[22];
    const float* h_w1     = (const float*)d_in[23];
    const float* h_b1     = (const float*)d_in[24];
    const float* h_w2     = (const float*)d_in[25];
    const float* h_b2     = (const float*)d_in[26];

    // ---- workspace carve ----
    char* wp = (char*)d_ws;
    float* x    = (float*)wp;  wp += (size_t)NTOK * DD * 4;    // residual stream f32
    float* buf1 = (float*)wp;  wp += (size_t)NTOK * DD * 4;    // pre-LN f32 GEMM outputs
    u16*   xb   = (u16*)wp;    wp += (size_t)NTOK * DD * 2;    // bf16 mirror of GEMM input
    u16*   o_b  = (u16*)wp;    wp += (size_t)NTOK * DD * 2;    // bf16 attention out
    u16*   wide = (u16*)wp;    wp += (size_t)NTOK * DFF * 2;   // bf16 wide activations
    u16*   w1t  = (u16*)wp;    wp += (size_t)NM * DIN * DD * 2;
    u16*   w2t  = (u16*)wp;    wp += (size_t)NM * DD * DIN * 2;
    u16*   qkvt = (u16*)wp;    wp += (size_t)NA * 3 * DD * DD * 2;
    u16*   aot  = (u16*)wp;    wp += (size_t)NA * DD * DD * 2;
    u16*   f1t  = (u16*)wp;    wp += (size_t)NA * DFF * DD * 2;
    u16*   f2t  = (u16*)wp;    wp += (size_t)NA * DD * DFF * 2;

    dim3 blk(128);
    auto cvt = [&](const float* W, u16* Wt, int K, int N) {
        int total = K * N;
        cer_wt_kernel<<<(total + 255) / 256, 256, 0, stream>>>(W, Wt, K, N);
    };

    // pre-convert + transpose all weights to bf16 W^T[N][K]
    for (int i = 0; i < NM; ++i) {
        cvt(m_W1 + (size_t)i * DD * DIN, w1t + (size_t)i * DIN * DD, DD, DIN);
        cvt(m_W2 + (size_t)i * DIN * DD, w2t + (size_t)i * DD * DIN, DIN, DD);
    }
    for (int i = 0; i < NA; ++i) {
        cvt(a_qkv_w + (size_t)i * DD * 3 * DD, qkvt + (size_t)i * 3 * DD * DD, DD, 3 * DD);
        cvt(a_out_w + (size_t)i * DD * DD,     aot  + (size_t)i * DD * DD,     DD, DD);
        cvt(a_ff_w1 + (size_t)i * DD * DFF,    f1t  + (size_t)i * DFF * DD,    DD, DFF);
        cvt(a_ff_w2 + (size_t)i * DFF * DD,    f2t  + (size_t)i * DD * DFF,    DFF, DD);
    }

    // x = emb[ids] + pos
    cer_embed_kernel<<<NTOK, blk, 0, stream>>>(ids, emb, pos_emb, x);

    // MLP blocks
    for (int i = 0; i < NM; ++i) {
        cer_ln_kernel<<<NTOK, blk, 0, stream>>>(x, nullptr, m_ln_w + i * DD, m_ln_b + i * DD,
                                                nullptr, xb);
        cer_gemm_kernel<<<dim3(NTOK / 64, DIN / 64), blk, 0, stream>>>(
            xb, w1t + (size_t)i * DIN * DD, m_b1 + i * DIN, nullptr,
            nullptr, wide, NTOK, DD, DIN, ACT_SILU);
        cer_gemm_kernel<<<dim3(NTOK / 64, DD / 64), blk, 0, stream>>>(
            wide, w2t + (size_t)i * DD * DIN, m_b2 + i * DD, x,
            x, (i == NM - 1) ? xb : nullptr, NTOK, DIN, DD, ACT_NONE);
    }

    // Attention blocks
    for (int i = 0; i < NA; ++i) {
        cer_gemm_kernel<<<dim3(NTOK / 64, (3 * DD) / 64), blk, 0, stream>>>(
            xb, qkvt + (size_t)i * 3 * DD * DD, a_qkv_b + i * 3 * DD, nullptr,
            nullptr, wide, NTOK, DD, 3 * DD, ACT_NONE);
        cer_attn_kernel<<<dim3(SS / 64, HH, BB), blk, 0, stream>>>(wide, o_b);
        cer_gemm_kernel<<<dim3(NTOK / 64, DD / 64), blk, 0, stream>>>(
            o_b, aot + (size_t)i * DD * DD, a_out_b + i * DD, nullptr,
            buf1, nullptr, NTOK, DD, DD, ACT_NONE);
        cer_ln_kernel<<<NTOK, blk, 0, stream>>>(buf1, x, a_ln1_w + i * DD, a_ln1_b + i * DD,
                                                x, xb);
        cer_gemm_kernel<<<dim3(NTOK / 64, DFF / 64), blk, 0, stream>>>(
            xb, f1t + (size_t)i * DFF * DD, a_ff_b1 + i * DFF, nullptr,
            nullptr, wide, NTOK, DD, DFF, ACT_RELU);
        cer_gemm_kernel<<<dim3(NTOK / 64, DD / 64), blk, 0, stream>>>(
            wide, f2t + (size_t)i * DD * DFF, a_ff_b2 + i * DD, nullptr,
            buf1, nullptr, NTOK, DFF, DD, ACT_NONE);
        cer_ln_kernel<<<NTOK, blk, 0, stream>>>(buf1, x, a_ln2_w + i * DD, a_ln2_b + i * DD,
                                                x, xb);
    }

    // final LN + head
    cer_ln_kernel<<<NTOK, blk, 0, stream>>>(x, nullptr, fn_w, fn_b, buf1, nullptr);
    cer_head_kernel<<<BB, blk, 0, stream>>>(buf1, h_w1, h_b1, h_w2, h_b2, (float*)d_out);
}
